// CausalSelfAttention_87170656239617
// MI455X (gfx1250) — compile-verified, matrix operand order checked by reading, not run on hardware
//
#include <hip/hip_runtime.h>
#include <hip/hip_bf16.h>
#include <stdint.h>

// ---------------------------------------------------------------------------
// CDNA5 (gfx1250) causal self-attention, bf16 WMMA pipeline + TDM staging.
// ---------------------------------------------------------------------------

typedef __attribute__((ext_vector_type(16))) __bf16 v16bf;
typedef __attribute__((ext_vector_type(8)))  __bf16 v8bf;
typedef __attribute__((ext_vector_type(8)))  float  v8f;
typedef __attribute__((ext_vector_type(4)))  float  v4f;
typedef __attribute__((ext_vector_type(4)))  unsigned int u32x4;
typedef __attribute__((ext_vector_type(8)))  int    i32x8;
typedef __attribute__((ext_vector_type(4)))  int    i32x4;

#define DEVINL __device__ __forceinline__

// Problem constants
#define BB   4
#define TT   2048
#define CC   1024
#define NH   16
#define NKV  4
#define HD   64
#define MM   (BB * TT)   // 8192 token rows

#if defined(__has_builtin)
#if __has_builtin(__builtin_amdgcn_tensor_load_to_lds) && \
    __has_builtin(__builtin_amdgcn_s_wait_tensorcnt)
#define USE_TDM 1
#endif
#endif
#ifndef USE_TDM
#define USE_TDM 0
#endif

DEVINL __bf16 f2bf(float f) {
  union { float f; uint32_t u; } x; x.f = f;
  uint32_t r = x.u + 0x7fffu + ((x.u >> 16) & 1u);   // RNE
  union { uint16_t s; __bf16 b; } y; y.s = (uint16_t)(r >> 16);
  return y.b;
}

// Load a 16-bit WMMA fragment (A: row=lane%16, or B: col=lane%16) from LDS.
//   elems 0..7  -> K = kbase + half*8 + e
//   elems 8..15 -> K = kbase + 16 + half*8 + (e-8)
DEVINL v16bf load_frag(const __bf16* rowbase, int kbase, int half) {
  union { v16bf v; v8bf h[2]; } r;
  r.h[0] = *(const v8bf*)(rowbase + kbase + half * 8);
  r.h[1] = *(const v8bf*)(rowbase + kbase + 16 + half * 8);
  return r.v;
}

DEVINL v8f wmma_bf16(v16bf a, v16bf b, v8f c) {
  return __builtin_amdgcn_wmma_f32_16x16x32_bf16(
      /*neg_a=*/false, a, /*neg_b=*/false, b,
      /*c_mod=*/(short)0, c, /*reuse_a=*/false, /*reuse_b=*/false);
}

#if USE_TDM
// ---- Tensor DMA descriptor builders (CDNA5 D#, ISA 08_async_tensor §8) ----
// Group 0: [1:0]=count=1, [63:32]=lds_addr, [120:64]=global_addr, type=2.
DEVINL u32x4 tdm_g0(uint32_t lds_addr, uint64_t gaddr) {
  u32x4 g;
  g[0] = 1u;                                           // count=1, user desc
  g[1] = lds_addr;                                     // LDS byte address
  g[2] = (uint32_t)gaddr;                              // global_addr[31:0]
  g[3] = (uint32_t)((gaddr >> 32) & 0x01FFFFFFu)       // global_addr[56:32]
         | 0x80000000u;                                // type=2 ("image")
  return g;
}
// Group 1: data_size=1 (2 bytes), workgroup_mask=0 (not in a cluster),
// tensor_dim0/1, tile_dim0/1 (tile_dim2=0), tensor_dim0_stride.
DEVINL i32x8 tdm_g1(uint32_t tensor_d0, uint32_t tensor_d1,
                    uint32_t tile_d0, uint32_t tile_d1, uint32_t stride0) {
  uint32_t w0 = (1u << 16);                            // data_size=2B
  uint32_t w1 = (tensor_d0 & 0xffffu) << 16;           // tensor_dim0[15:0]
  uint32_t w2 = ((tensor_d0 >> 16) & 0xffffu) |        // tensor_dim0[31:16]
                ((tensor_d1 & 0xffffu) << 16);         // tensor_dim1[15:0]
  uint32_t w3 = ((tensor_d1 >> 16) & 0xffffu) |        // tensor_dim1[31:16]
                ((tile_d0 & 0xffffu) << 16);           // tile_dim0
  uint32_t w4 = tile_d1 & 0xffffu;                     // tile_dim1 (dim2=0)
  uint32_t w5 = stride0;                               // dim0_stride[31:0]
  i32x8 g;
  g[0] = (int)w0; g[1] = (int)w1; g[2] = (int)w2; g[3] = (int)w3;
  g[4] = (int)w4; g[5] = (int)w5; g[6] = 0; g[7] = 0;
  return g;
}
// 2D tile load: rows x cols (bf16), row-major source with given row stride.
DEVINL void tdm_load_2d(const __bf16* gsrc, const __bf16* lds_dst,
                        uint32_t cols, uint32_t total_rows,
                        uint32_t tile_rows, uint32_t row_stride) {
  u32x4 g0 = tdm_g0((uint32_t)(uintptr_t)lds_dst, (uint64_t)(uintptr_t)gsrc);
  i32x8 g1 = tdm_g1(cols, total_rows, cols, tile_rows, row_stride);
  i32x4 z4 = {};
  i32x8 z8 = {};
  // 6-arg toolchain form: (g0, g1, g2, g3, g_extra, cpol)
  __builtin_amdgcn_tensor_load_to_lds(g0, g1, z4, z4, z8, 0);
}
#endif  // USE_TDM

// ---------------------------------------------------------------------------
// GEMM: C[M,N] (fp32) = A[M,K] * B[N,K]^T      (B is fp32 weights)
// Block tile 128x128, K-step 32, 256 threads = 8 waves (4x2), each wave a
// 32x64 subtile = 2x4 WMMA accumulators.
// ---------------------------------------------------------------------------
template <bool A_IS_BF16>
__global__ __launch_bounds__(256)
void gemm_wmma(const void* __restrict__ Aptr, const float* __restrict__ B,
               float* __restrict__ C, int M, int N, int K) {
  __shared__ __align__(16) __bf16 As[128 * 32];
  __shared__ __align__(16) __bf16 Bs[128 * 32];

  const int tid  = threadIdx.x;
  const int lane = tid & 31, wave = tid >> 5;
  const int wm = wave >> 1, wn = wave & 1;      // 4x2 wave grid
  const int half = lane >> 4, l16 = lane & 15;
  const int m0 = blockIdx.y * 128, n0 = blockIdx.x * 128;

  v8f acc[2][4];
#pragma unroll
  for (int i = 0; i < 2; ++i)
#pragma unroll
    for (int j = 0; j < 4; ++j) acc[i][j] = {};

  const int rowS = tid >> 1;            // 0..127
  const int colS = (tid & 1) * 16;      // 0 or 16

  for (int k0 = 0; k0 < K; k0 += 32) {
    // ---- stage A tile (128x32) as bf16 ----
    if (A_IS_BF16) {
      const __bf16* A = (const __bf16*)Aptr;
      const v8bf* s = (const v8bf*)(A + (size_t)(m0 + rowS) * K + k0 + colS);
      v8bf* d = (v8bf*)(As + rowS * 32 + colS);
      d[0] = s[0]; d[1] = s[1];
      __builtin_prefetch(s + 4, 0, 0);
    } else {
      const float* A = (const float*)Aptr;
      const float* s = A + (size_t)(m0 + rowS) * K + k0 + colS;
      __builtin_prefetch(s + 32, 0, 0);
#pragma unroll
      for (int i = 0; i < 16; i += 4) {
        v4f f = *(const v4f*)(s + i);
        As[rowS * 32 + colS + i + 0] = f2bf(f[0]);
        As[rowS * 32 + colS + i + 1] = f2bf(f[1]);
        As[rowS * 32 + colS + i + 2] = f2bf(f[2]);
        As[rowS * 32 + colS + i + 3] = f2bf(f[3]);
      }
    }
    // ---- stage B tile (128 N-rows x 32 K) as bf16, [n][k] layout ----
    {
      const float* s = B + (size_t)(n0 + rowS) * K + k0 + colS;
      __builtin_prefetch(s + 32, 0, 0);
#pragma unroll
      for (int i = 0; i < 16; i += 4) {
        v4f f = *(const v4f*)(s + i);
        Bs[rowS * 32 + colS + i + 0] = f2bf(f[0]);
        Bs[rowS * 32 + colS + i + 1] = f2bf(f[1]);
        Bs[rowS * 32 + colS + i + 2] = f2bf(f[2]);
        Bs[rowS * 32 + colS + i + 3] = f2bf(f[3]);
      }
    }
    __syncthreads();

    v16bf af[2], bf[4];
#pragma unroll
    for (int mt = 0; mt < 2; ++mt)
      af[mt] = load_frag(As + (wm * 32 + mt * 16 + l16) * 32, 0, half);
#pragma unroll
    for (int nt = 0; nt < 4; ++nt)
      bf[nt] = load_frag(Bs + (wn * 64 + nt * 16 + l16) * 32, 0, half);
#pragma unroll
    for (int mt = 0; mt < 2; ++mt)
#pragma unroll
      for (int nt = 0; nt < 4; ++nt)
        acc[mt][nt] = wmma_bf16(af[mt], bf[nt], acc[mt][nt]);
    __syncthreads();
  }

  // C layout: vgpr j -> row = j + 8*half, col = l16
#pragma unroll
  for (int mt = 0; mt < 2; ++mt)
#pragma unroll
    for (int nt = 0; nt < 4; ++nt)
#pragma unroll
      for (int j = 0; j < 8; ++j) {
        int m = m0 + wm * 32 + mt * 16 + half * 8 + j;
        int n = n0 + wn * 64 + nt * 16 + l16;
        C[(size_t)m * N + n] = acc[mt][nt][j];
      }
}

// ---------------------------------------------------------------------------
// RoPE + QK RMS-norm + bf16 cast.  One wave per (token, head) row of 64;
// lane i owns the interleaved pair (2i, 2i+1).
// in : fp32 [B*T, nh*64]   out: bf16 [b][nh][t][64]
// ---------------------------------------------------------------------------
__global__ __launch_bounds__(256)
void rope_norm_cast(const float* __restrict__ in, __bf16* __restrict__ out,
                    int nh) {
  const int lane = threadIdx.x & 31;
  const int row  = blockIdx.x * 8 + (threadIdx.x >> 5);  // over B*T*nh
  const int h  = row % nh;
  const int bt = row / nh;
  const int t  = bt % TT;
  const int b  = bt / TT;

  const float* p = in + (size_t)row * HD;
  float xe = p[2 * lane], xo = p[2 * lane + 1];

  // inv_freq = theta^(-2i/64) = exp(-i * ln(10000)/32)
  float fr = (float)t * __expf(-(float)lane * 0.28782313792f);
  float s, c;
  __sincosf(fr, &s, &c);
  float oe = xe * c - xo * s;
  float oo = xe * s + xo * c;

  float ss = oe * oe + oo * oo;
#pragma unroll
  for (int m = 16; m >= 1; m >>= 1) ss += __shfl_xor(ss, m, 32);
  float inv = 1.0f / sqrtf(ss * (1.0f / 64.0f) + 1e-6f);

  __bf16* q = out + (((size_t)(b * nh + h) * TT) + t) * HD;
  q[2 * lane]     = f2bf(oe * inv);
  q[2 * lane + 1] = f2bf(oo * inv);
}

// V: fp32 [b][t][kv][64]  ->  bf16 [b][kv][t][64]
__global__ __launch_bounds__(256)
void cast_v(const float* __restrict__ in, __bf16* __restrict__ out) {
  size_t i = (size_t)blockIdx.x * 256 + threadIdx.x;  // over B*T*NKV*64
  int d = (int)(i & 63);
  size_t r = i >> 6;
  int kv = (int)(r % NKV);
  size_t bt = r / NKV;
  int t = (int)(bt % TT);
  int b = (int)(bt / TT);
  out[(((size_t)(b * NKV + kv) * TT) + t) * HD + d] = f2bf(in[i]);
}

// ---------------------------------------------------------------------------
// Causal flash attention.  Block = (b, h, 128 q-rows); 8 waves x 16 rows.
// Q bf16 [b][h][t][64], K bf16 [b][kv][t][64], V bf16 [b][kv][t][64].
// Y bf16 [b][t][h*64+d]  (A-operand layout for the final projection GEMM).
// Q block and K tiles are staged LDS-side by the Tensor Data Mover
// (one DMA per block, issued by wave 0, tracked with TENSORcnt).
// ---------------------------------------------------------------------------
__global__ __launch_bounds__(256)
void flash_attn(const __bf16* __restrict__ Q, const __bf16* __restrict__ Kp,
                const __bf16* __restrict__ V, __bf16* __restrict__ Y) {
  __shared__ __align__(16) __bf16 Qs[128 * 64];   // [qrow][d]
  __shared__ __align__(16) __bf16 Ks[32 * 64];    // [key][d]
  __shared__ __align__(16) __bf16 VsT[64 * 32];   // [d][key]
  __shared__ __align__(16) __bf16 Ps[8][16 * 32]; // per-wave P scratch

  const int tid  = threadIdx.x;
  const int lane = tid & 31, wave = tid >> 5;
  const int half = lane >> 4, l16 = lane & 15;
  const int b = blockIdx.z, h = blockIdx.y, q0 = blockIdx.x * 128;
  const int kvh = h / (NH / NKV);

  const __bf16* qg = Q  + ((size_t)(b * NH  + h)   * TT + q0) * HD;
  const __bf16* kg = Kp + ((size_t)(b * NKV + kvh) * TT) * HD;
  const __bf16* vg = V  + ((size_t)(b * NKV + kvh) * TT) * HD;

  // ---- stage Q block (128 x 64) ----
#if USE_TDM
  if (wave == 0) {
    tdm_load_2d(qg, Qs, /*cols=*/HD, /*total_rows=*/TT - q0,
                /*tile_rows=*/128, /*row_stride=*/HD);
    __builtin_amdgcn_s_wait_tensorcnt(0);
  }
#else
  {
    int r = tid >> 1, c = (tid & 1) * 32;
    const v8bf* s = (const v8bf*)(qg + (size_t)r * HD + c);
    v8bf* d = (v8bf*)(Qs + r * HD + c);
    d[0] = s[0]; d[1] = s[1]; d[2] = s[2]; d[3] = s[3];
  }
#endif

  v8f acc_o[4];
#pragma unroll
  for (int nt = 0; nt < 4; ++nt) acc_o[nt] = {};
  float mrow[8], lrow[8];
#pragma unroll
  for (int j = 0; j < 8; ++j) { mrow[j] = -1e30f; lrow[j] = 0.0f; }

  const float scale = 0.125f;          // 1/sqrt(64)
  const int qrow_base = q0 + wave * 16;
  const int nk = q0 / 32 + 4;          // key tiles covering keys <= q0+127

  for (int kt = 0; kt < nk; ++kt) {
    const int k0 = kt * 32;
    __syncthreads();   // all waves done reading Ks/VsT from previous iter
    // ---- stage K tile (32 x 64) via TDM ----
#if USE_TDM
    if (wave == 0) {
      tdm_load_2d(kg + (size_t)k0 * HD, Ks, /*cols=*/HD,
                  /*total_rows=*/TT - k0, /*tile_rows=*/32,
                  /*row_stride=*/HD);
    }
#else
    {
      int r = tid >> 3, cb = (tid & 7) * 8;
      *(v8bf*)(Ks + r * HD + cb) =
          *(const v8bf*)(kg + (size_t)(k0 + r) * HD + cb);
    }
#endif
    // ---- stage V tile transposed -> [d][key] (TDM cannot transpose) ----
    {
      int tk = tid & 31, db = (tid >> 5) * 8;
      v8bf vv = *(const v8bf*)(vg + (size_t)(k0 + tk) * HD + db);
#pragma unroll
      for (int i = 0; i < 8; ++i) VsT[(db + i) * 32 + tk] = vv[i];
    }
#if USE_TDM
    if (wave == 0) __builtin_amdgcn_s_wait_tensorcnt(0);
#endif
    __syncthreads();

    // scores: S = Q(16x64) . K^T(64x32)  -> two 16x16 accumulators
    v8f s0 = {}, s1 = {};
#pragma unroll
    for (int dk = 0; dk < HD; dk += 32) {
      v16bf aq = load_frag(Qs + (wave * 16 + l16) * HD, dk, half);
      v16bf b0 = load_frag(Ks + (l16) * HD, dk, half);
      v16bf b1 = load_frag(Ks + (16 + l16) * HD, dk, half);
      s0 = wmma_bf16(aq, b0, s0);
      s1 = wmma_bf16(aq, b1, s1);
    }

    // online softmax (rows replicated across the 16-lane half group)
#pragma unroll
    for (int j = 0; j < 8; ++j) {
      int qrow = qrow_base + half * 8 + j;
      float v0 = s0[j] * scale, v1 = s1[j] * scale;
      if (k0 + l16 > qrow)      v0 = -1e30f;
      if (k0 + 16 + l16 > qrow) v1 = -1e30f;
      float rm = fmaxf(v0, v1);
#pragma unroll
      for (int m = 8; m >= 1; m >>= 1) rm = fmaxf(rm, __shfl_xor(rm, m, 32));
      float mnew = fmaxf(mrow[j], rm);
      float p0 = __expf(v0 - mnew), p1 = __expf(v1 - mnew);
      float rs = p0 + p1;
#pragma unroll
      for (int m = 8; m >= 1; m >>= 1) rs += __shfl_xor(rs, m, 32);
      float alpha = __expf(mrow[j] - mnew);
      mrow[j] = mnew;
      lrow[j] = lrow[j] * alpha + rs;
#pragma unroll
      for (int nt = 0; nt < 4; ++nt) acc_o[nt][j] *= alpha;
      __bf16* pr = &Ps[wave][(half * 8 + j) * 32];
      pr[l16]      = f2bf(p0);
      pr[16 + l16] = f2bf(p1);
    }

    // O += P(16x32) . V(32x64)
    v16bf pf = load_frag(&Ps[wave][l16 * 32], 0, half);
#pragma unroll
    for (int nt = 0; nt < 4; ++nt) {
      v16bf vf = load_frag(VsT + (nt * 16 + l16) * 32, 0, half);
      acc_o[nt] = wmma_bf16(pf, vf, acc_o[nt]);
    }
  }

  // epilogue: normalize and write Y[b][t][h*64+d]
#pragma unroll
  for (int nt = 0; nt < 4; ++nt)
#pragma unroll
    for (int j = 0; j < 8; ++j) {
      int qrow = qrow_base + half * 8 + j;
      float o = acc_o[nt][j] / lrow[j];
      Y[((size_t)(b * TT + qrow)) * (NH * HD) + h * HD + nt * 16 + l16] =
          f2bf(o);
    }
}

// ---------------------------------------------------------------------------
extern "C" void kernel_launch(void* const* d_in, const int* in_sizes, int n_in,
                              void* d_out, int out_size, void* d_ws,
                              size_t ws_size, hipStream_t stream) {
  (void)in_sizes; (void)n_in; (void)out_size; (void)ws_size;
  const float* x  = (const float*)d_in[0];
  const float* wq = (const float*)d_in[1];
  const float* wk = (const float*)d_in[2];
  const float* wv = (const float*)d_in[3];
  const float* wo = (const float*)d_in[4];
  float* out = (float*)d_out;

  char* ws = (char*)d_ws;
  size_t off = 0;
  auto take = [&](size_t bytes) {
    char* p = ws + off;
    off += (bytes + 255) & ~(size_t)255;
    return p;
  };
  float*  qtmp = (float*)take((size_t)MM * 1024 * 4);
  float*  ktmp = (float*)take((size_t)MM * 256 * 4);
  float*  vtmp = (float*)take((size_t)MM * 256 * 4);
  __bf16* qbf  = (__bf16*)take((size_t)MM * 1024 * 2);
  __bf16* kbf  = (__bf16*)take((size_t)MM * 256 * 2);
  __bf16* vbf  = (__bf16*)take((size_t)MM * 256 * 2);
  __bf16* ybf  = (__bf16*)qtmp;  // reuse: qtmp fully consumed before attention

  dim3 blk(256);
  // Q/K/V projections (fp32 A staged to bf16)
  gemm_wmma<false><<<dim3(1024 / 128, MM / 128), blk, 0, stream>>>(
      x, wq, qtmp, MM, 1024, 1024);
  gemm_wmma<false><<<dim3(256 / 128, MM / 128), blk, 0, stream>>>(
      x, wk, ktmp, MM, 256, 1024);
  gemm_wmma<false><<<dim3(256 / 128, MM / 128), blk, 0, stream>>>(
      x, wv, vtmp, MM, 256, 1024);
  // RoPE + RMS norm + relayout to bf16
  rope_norm_cast<<<(MM * NH) / 8, blk, 0, stream>>>(qtmp, qbf, NH);
  rope_norm_cast<<<(MM * NKV) / 8, blk, 0, stream>>>(ktmp, kbf, NKV);
  cast_v<<<(MM * NKV * HD) / 256, blk, 0, stream>>>(vtmp, vbf);
  // causal flash attention
  flash_attn<<<dim3(TT / 128, NH, BB), blk, 0, stream>>>(qbf, kbf, vbf, ybf);
  // output projection (bf16 A)
  gemm_wmma<true><<<dim3(1024 / 128, MM / 128), blk, 0, stream>>>(
      ybf, wo, out, MM, 1024, 1024);
}
